// HMNeRF_50448685858904
// MI455X (gfx1250) — matrix operand order repaired and checked by reading, not compiled, not run on hardware
//
#include <hip/hip_runtime.h>
#include <cstdint>
#include <cstddef>

// ---------------------------------------------------------------------------
// MI455X (gfx1250) HMNeRF forward.
//   - big 256-wide MLP on v_wmma_f32_16x16x32_f16 (f16 in, f32 accum)
//   - weights fp32->f16 once into workspace (K padded to 32-multiples)
//   - per-block LDS weight staging (double-buffered chunks, async-to-LDS)
//     so the 8 waves share one L2 read of each weight matrix
//   - single LDS activation buffer: each wave preloads all A-fragments into
//     VGPRs at layer start, so outputs may overwrite inputs in place
//   - small per-video MLP on VALU (per-row weight gather), partials in ws
// ---------------------------------------------------------------------------

typedef __attribute__((ext_vector_type(16))) _Float16 v16h;
typedef __attribute__((ext_vector_type(8)))  float    v8f;
typedef __attribute__((ext_vector_type(4)))  int      v4i;

union Frag16 {
  v16h      v;
  _Float16  h[16];
  float4    f4[2];
};

#define HS 264            // LDS activation row stride in halves (256 + pad)
#define XSTR 90           // x row stride (63 xyz + 27 xdir)
#define WSLOT 11520       // weight chunk slot: 32 rows * (352+8) halves
#define NTHR 256

#define HAVE_ASYNC_LDS __has_builtin(__builtin_amdgcn_global_load_async_to_lds_b128)

#if HAVE_ASYNC_LDS
// Signature (from clang diagnostic): (v4i __device__*, v4i __shared__*, imm, imm)
__device__ __forceinline__ void async_cp16(const _Float16* g, _Float16* l) {
  __builtin_amdgcn_global_load_async_to_lds_b128(
      (__attribute__((address_space(1))) v4i*)(g),
      (__attribute__((address_space(3))) v4i*)(l), 0, 0);
}
__device__ __forceinline__ void async_wait0() {
#if __has_builtin(__builtin_amdgcn_s_wait_asynccnt)
  __builtin_amdgcn_s_wait_asynccnt(0);
#else
  asm volatile("s_wait_asynccnt 0x0" ::: "memory");
#endif
}
#endif

// ---------------------------------------------------------------------------
// fp32 -> f16 weight conversion with K padding (and optional zero-column
// insertion at `ins` for the skip layer so K offsets stay 32-aligned).
// ---------------------------------------------------------------------------
__global__ __launch_bounds__(256) void cvt_pad(const float* __restrict__ src,
                                               _Float16* __restrict__ dst,
                                               int rows, int scols, int dcols,
                                               int ins) {
  int i = blockIdx.x * blockDim.x + threadIdx.x;
  if (i >= rows * dcols) return;
  int r = i / dcols, c = i % dcols;
  int sc = c;
  if (ins >= 0) {
    if (c == ins) { dst[i] = (_Float16)0.f; return; }
    if (c > ins) sc = c - 1;
  }
  float v = (sc < scols) ? src[(size_t)r * scols + sc] : 0.f;
  dst[i] = (_Float16)v;
}

// ---------------------------------------------------------------------------
// Small per-video MLP (VALU). Writes [sigmoid(rgb0..2), sigma] partials.
// ---------------------------------------------------------------------------
struct SmallParams {
  const float* x; const int* vid;
  const float *W0, *b0, *W1, *b1, *Wf, *bf, *Wd, *bd, *Ws, *bs, *Wr, *br;
  float* part;
};

__global__ __launch_bounds__(256) void mnerf_kernel(SmallParams P, int n_total) {
  int n = blockIdx.x * blockDim.x + threadIdx.x;
  if (n >= n_total) return;
  const int v = P.vid[n];
  const float* xr = P.x + (size_t)n * XSTR;

  float h[64], g[64];
  { const float* Wv = P.W0 + (size_t)v * 64 * 63; const float* bv = P.b0 + v * 64;
    for (int o = 0; o < 64; ++o) { float a = bv[o]; const float* w = Wv + o * 63;
      for (int k = 0; k < 63; ++k) a += xr[k] * w[k];
      h[o] = fmaxf(a, 0.f); } }
  { const float* Wv = P.W1 + (size_t)v * 4096; const float* bv = P.b1 + v * 64;
    for (int o = 0; o < 64; ++o) { float a = bv[o]; const float* w = Wv + o * 64;
      for (int k = 0; k < 64; ++k) a += h[k] * w[k];
      g[o] = fmaxf(a, 0.f); } }
  float sigma = P.bs[v];
  { const float* w = P.Ws + v * 64;
    for (int k = 0; k < 64; ++k) sigma += g[k] * w[k]; }
  float f[64];
  { const float* Wv = P.Wf + (size_t)v * 4096; const float* bv = P.bf + v * 64;
    for (int o = 0; o < 64; ++o) { float a = bv[o]; const float* w = Wv + o * 64;
      for (int k = 0; k < 64; ++k) a += g[k] * w[k];
      f[o] = a; } }
  float dd[32];
  { const float* Wv = P.Wd + (size_t)v * 32 * 91; const float* bv = P.bd + v * 32;
    for (int o = 0; o < 32; ++o) { float a = bv[o]; const float* w = Wv + o * 91;
      for (int k = 0; k < 64; ++k) a += f[k] * w[k];
      for (int j = 0; j < 27; ++j) a += xr[63 + j] * w[64 + j];
      dd[o] = fmaxf(a, 0.f); } }
  float* po = P.part + (size_t)n * 4;
  { const float* Wv = P.Wr + v * 96; const float* bv = P.br + v * 3;
    for (int c = 0; c < 3; ++c) { float a = bv[c]; const float* w = Wv + c * 32;
      for (int k = 0; k < 32; ++k) a += dd[k] * w[k];
      po[c] = 1.f / (1.f + __expf(-a)); } }   // reference sigmoids rgb here
  po[3] = sigma;
}

// ---------------------------------------------------------------------------
// A 16x32 f16 fragment per the gfx1250 layout:
//   lane<16 : halves 0..7 = K k0..k0+7,  8..15 = K k0+16..k0+23 (row = lane)
//   lane>=16: halves 0..7 = K k0+8..15,  8..15 = K k0+24..31    (row = lane-16)
// i.e. two 16B chunks at (k0 + 8*hi) and (k0 + 16 + 8*hi); p is lane-adjusted.
// ---------------------------------------------------------------------------
__device__ __forceinline__ Frag16 ld_afrag(const _Float16* p, int kofs) {
  Frag16 f;
  f.f4[0] = *(const float4*)(p + kofs);
  f.f4[1] = *(const float4*)(p + kofs + 16);
  return f;
}

// Cooperative stage of one weight chunk (32 output rows x KS halves) into an
// LDS slot with +8-half row padding (bank spread). 16B granules per lane.
template <int KS>
__device__ __forceinline__ void stage_chunk(const _Float16* __restrict__ Wh,
                                            _Float16* __restrict__ dst,
                                            int c, int tid) {
  constexpr int GPR = KS / 8;       // 16B granules per row
  constexpr int TOTAL = 32 * GPR;
  const _Float16* src = Wh + (size_t)c * 32 * KS;
  for (int i = tid; i < TOTAL; i += NTHR) {
    const int r = i / GPR, g = i - r * GPR;
    const _Float16* s = src + r * KS + g * 8;
    _Float16* d = dst + r * (KS + 8) + g * 8;
#if HAVE_ASYNC_LDS
    async_cp16(s, d);
#else
    *(float4*)d = *(const float4*)s;
#endif
  }
}

// out[m][n] = act( sum_k A[m][k]*W[n][k] + b[n] ), weights staged through LDS.
// B fragment from LDS slot: column = local col, two 8-half K chunks at
// k0 + 8*(lane/16) and k0 + 16 + 8*(lane/16).  C/D layout: vgpr r of lane L -> (M=r+8*(L>=16), N=L%16).
// Chunks of 2 N-tiles are double-buffered: stage c+1 while computing c.
template <int NKT, int NNT, bool RELU, int KS>
__device__ __forceinline__ void wmma_layer_lds(const Frag16* afr,
                                               const _Float16* __restrict__ Wh,
                                               const float* __restrict__ bias,
                                               _Float16* __restrict__ hout,
                                               _Float16* __restrict__ wbuf,
                                               int lr0, int lane, int tid) {
  constexpr int WKS = KS + 8;
  constexpr int NCH = NNT / 2;
  const int nl = lane & 15;
  const int hi = lane >> 4;

  stage_chunk<KS>(Wh, wbuf, 0, tid);
#if HAVE_ASYNC_LDS
  async_wait0();
#endif
  __syncthreads();

  for (int c = 0; c < NCH; ++c) {
    const _Float16* slot = wbuf + (c & 1) * WSLOT;
    if (c + 1 < NCH) stage_chunk<KS>(Wh, wbuf + ((c + 1) & 1) * WSLOT, c + 1, tid);
#pragma unroll
    for (int t = 0; t < 2; ++t) {
      const int n0 = (2 * c + t) * 16;
      const _Float16* wp = slot + (t * 16 + nl) * WKS + hi * 8;
      const float bv = bias[n0 + nl];
      v8f acc = {};
#pragma unroll
      for (int kt = 0; kt < NKT; ++kt) {
        Frag16 bf;
        bf.f4[0] = *(const float4*)(wp + kt * 32);
        bf.f4[1] = *(const float4*)(wp + kt * 32 + 16);
        acc = __builtin_amdgcn_wmma_f32_16x16x32_f16(
            false, afr[kt].v, false, bf.v, (short)0, acc, false, false);
      }
#pragma unroll
      for (int r = 0; r < 8; ++r) {
        float vv = acc[r] + bv;
        if (RELU) vv = fmaxf(vv, 0.f);
        hout[(lr0 + r + 8 * hi) * HS + n0 + nl] = (_Float16)vv;
      }
    }
#if HAVE_ASYNC_LDS
    async_wait0();
#endif
    __syncthreads();
  }
}

// ---------------------------------------------------------------------------
// Big MLP: 8 wave32s / block, 128 points / block, 16 rows per wave.
// Activations: ONE buffer — A-frags preloaded to VGPRs, outputs overwrite.
// ---------------------------------------------------------------------------
struct BigParams {
  const float* x; const int* vid; const float* code;
  const float *b0, *b1, *b2, *b3, *b4, *b5, *b6, *b7;
  const float *bf, *bd, *Ws, *bs, *Wr, *br;
  const _Float16 *W0h, *W1h, *W2h, *W3h, *W4h, *W5h, *W6h, *W7h, *Wfh, *Wdh;
  const float* part_small;
  float* out;
};

__global__ __launch_bounds__(256) void bignerf_kernel(BigParams P) {
  extern __shared__ _Float16 smem[];
  _Float16* inp_s = smem;                       // 128 x 96: [xyz|code|0]
  _Float16* hbuf  = smem + 128 * 96;            // 128 x HS activations
  _Float16* wbuf  = hbuf + 128 * HS;            // 2 x WSLOT weight chunks

  const int tid = threadIdx.x;
  const int wid = tid >> 5;
  const int lane = tid & 31;
  const int lr0 = wid * 16;                     // this wave's local row base
  const int row0 = blockIdx.x * 128 + lr0;      // global row base
  const int m = lane & 15;
  const int hi = lane >> 4;

  // Stage input (f32 -> f16); per-wave rows, and inp_s is never overwritten.
  for (int r = 0; r < 16; ++r) {
    const int row = row0 + r;
    const int v = P.vid[row];
    for (int c = lane; c < 96; c += 32) {
      float val = 0.f;
      if (c < 63)      val = P.x[(size_t)row * XSTR + c];
      else if (c < 95) val = P.code[v * 32 + (c - 63)];
      inp_s[(lr0 + r) * 96 + c] = (_Float16)val;
    }
  }

  const _Float16* aip = inp_s + (lr0 + m) * 96 + 8 * hi;
  const _Float16* ap  = hbuf  + (lr0 + m) * HS + 8 * hi;

  Frag16 a11[11];
  Frag16 a8[8];

  // L0: 96 -> 256
#pragma unroll
  for (int kt = 0; kt < 3; ++kt) a11[kt] = ld_afrag(aip, kt * 32);
  wmma_layer_lds<3, 16, true, 96>(a11, P.W0h, P.b0, hbuf, wbuf, lr0, lane, tid);
  // L1..L3
#pragma unroll
  for (int kt = 0; kt < 8; ++kt) a8[kt] = ld_afrag(ap, kt * 32);
  wmma_layer_lds<8, 16, true, 256>(a8, P.W1h, P.b1, hbuf, wbuf, lr0, lane, tid);
#pragma unroll
  for (int kt = 0; kt < 8; ++kt) a8[kt] = ld_afrag(ap, kt * 32);
  wmma_layer_lds<8, 16, true, 256>(a8, P.W2h, P.b2, hbuf, wbuf, lr0, lane, tid);
#pragma unroll
  for (int kt = 0; kt < 8; ++kt) a8[kt] = ld_afrag(ap, kt * 32);
  wmma_layer_lds<8, 16, true, 256>(a8, P.W3h, P.b3, hbuf, wbuf, lr0, lane, tid);
  // L4 (skip): [inp(96) | h(256)] vs W4h (stride 352, zero col inserted @95)
#pragma unroll
  for (int kt = 0; kt < 3; ++kt) a11[kt] = ld_afrag(aip, kt * 32);
#pragma unroll
  for (int kt = 0; kt < 8; ++kt) a11[3 + kt] = ld_afrag(ap, kt * 32);
  wmma_layer_lds<11, 16, true, 352>(a11, P.W4h, P.b4, hbuf, wbuf, lr0, lane, tid);
  // L5..L7
#pragma unroll
  for (int kt = 0; kt < 8; ++kt) a8[kt] = ld_afrag(ap, kt * 32);
  wmma_layer_lds<8, 16, true, 256>(a8, P.W5h, P.b5, hbuf, wbuf, lr0, lane, tid);
#pragma unroll
  for (int kt = 0; kt < 8; ++kt) a8[kt] = ld_afrag(ap, kt * 32);
  wmma_layer_lds<8, 16, true, 256>(a8, P.W6h, P.b6, hbuf, wbuf, lr0, lane, tid);
#pragma unroll
  for (int kt = 0; kt < 8; ++kt) a8[kt] = ld_afrag(ap, kt * 32);
  wmma_layer_lds<8, 16, true, 256>(a8, P.W7h, P.b7, hbuf, wbuf, lr0, lane, tid);

  // sigma head (1x256) on VALU; lanes 0..15 own one row each (reads own rows
  // of hbuf before Wf overwrites them — in-wave ordering guarantees this).
  float sigma = 0.f;
  if (lane < 16) {
    float acc = P.bs[0];
    const _Float16* hp = hbuf + (lr0 + lane) * HS;
    for (int k = 0; k < 256; ++k) acc += (float)hp[k] * P.Ws[k];
    sigma = acc;
  }

  // final = h @ Wf^T + bf (linear)
#pragma unroll
  for (int kt = 0; kt < 8; ++kt) a8[kt] = ld_afrag(ap, kt * 32);
  wmma_layer_lds<8, 16, false, 256>(a8, P.Wfh, P.bf, hbuf, wbuf, lr0, lane, tid);

  // d = relu([final(256) | xdir(27) | 0pad] @ Wd^T + bd) -> hbuf cols 0..127
  Frag16 a9[9];
#pragma unroll
  for (int kt = 0; kt < 8; ++kt) a9[kt] = ld_afrag(ap, kt * 32);
  {
    const float* xd = P.x + (size_t)(row0 + m) * XSTR + 63;
#pragma unroll
    for (int e = 0; e < 8; ++e) {
      const int j0 = 8 * hi + e;          // K = 256 + j0
      const int j1 = 16 + 8 * hi + e;     // K = 256 + j1
      a9[8].h[e]     = (_Float16)((j0 < 27) ? xd[j0] : 0.f);
      a9[8].h[e + 8] = (_Float16)((j1 < 27) ? xd[j1] : 0.f);
    }
  }
  wmma_layer_lds<9, 8, true, 288>(a9, P.Wdh, P.bd, hbuf, wbuf, lr0, lane, tid);

  // rgb head (3x128) + combine:  sigmoid( sigmoid(big) + sigmoid(small) )
  if (lane < 16) {
    const int row = row0 + lane;
    const _Float16* dp = hbuf + (lr0 + lane) * HS;
    const float* ps = P.part_small + (size_t)row * 4;
    float* po = P.out + (size_t)row * 4;
#pragma unroll
    for (int c = 0; c < 3; ++c) {
      float acc = P.br[c];
      const float* wr = P.Wr + c * 128;
      for (int k = 0; k < 128; ++k) acc += (float)dp[k] * wr[k];
      const float s = 1.f / (1.f + __expf(-acc)) + ps[c];
      po[c] = 1.f / (1.f + __expf(-s));
    }
    po[3] = sigma + ps[3];
  }
}

// ---------------------------------------------------------------------------
// Host launcher.  d_in order = setup_inputs() dict insertion order:
//   0:x 1:vidid  2..17: (W0,b0..W7,b7)  18:Wf 19:bf 20:Wd 21:bd 22:Ws 23:bs
//   24:Wr 25:br 26:code  27..38: mnerf (W0,b0,W1,b1,Wf,bf,Wd,bd,Ws,bs,Wr,br)
// ---------------------------------------------------------------------------
extern "C" void kernel_launch(void* const* d_in, const int* in_sizes, int n_in,
                              void* d_out, int out_size, void* d_ws,
                              size_t ws_size, hipStream_t stream) {
  (void)n_in; (void)out_size; (void)ws_size;
  const float* x  = (const float*)d_in[0];
  const int* vid  = (const int*)d_in[1];
  const float *W[8], *b[8];
  for (int i = 0; i < 8; ++i) {
    W[i] = (const float*)d_in[2 + 2 * i];
    b[i] = (const float*)d_in[3 + 2 * i];
  }
  const float* Wf = (const float*)d_in[18];
  const float* bf = (const float*)d_in[19];
  const float* Wd = (const float*)d_in[20];
  const float* bd = (const float*)d_in[21];
  const float* Ws = (const float*)d_in[22];
  const float* bs = (const float*)d_in[23];
  const float* Wr = (const float*)d_in[24];
  const float* br = (const float*)d_in[25];
  const float* code = (const float*)d_in[26];

  SmallParams SP;
  SP.x = x; SP.vid = vid;
  SP.W0 = (const float*)d_in[27]; SP.b0 = (const float*)d_in[28];
  SP.W1 = (const float*)d_in[29]; SP.b1 = (const float*)d_in[30];
  SP.Wf = (const float*)d_in[31]; SP.bf = (const float*)d_in[32];
  SP.Wd = (const float*)d_in[33]; SP.bd = (const float*)d_in[34];
  SP.Ws = (const float*)d_in[35]; SP.bs = (const float*)d_in[36];
  SP.Wr = (const float*)d_in[37]; SP.br = (const float*)d_in[38];

  const int N = in_sizes[1];  // number of points (vidid count)

  // Workspace carve (f16 weight copies, then f32 small-net partials).
  _Float16* wb = (_Float16*)d_ws;
  size_t o = 0;
  _Float16* W0h = wb + o; o += 256 * 96;
  _Float16* W1h = wb + o; o += 256 * 256;
  _Float16* W2h = wb + o; o += 256 * 256;
  _Float16* W3h = wb + o; o += 256 * 256;
  _Float16* W4h = wb + o; o += 256 * 352;
  _Float16* W5h = wb + o; o += 256 * 256;
  _Float16* W6h = wb + o; o += 256 * 256;
  _Float16* W7h = wb + o; o += 256 * 256;
  _Float16* Wfh = wb + o; o += 256 * 256;
  _Float16* Wdh = wb + o; o += 128 * 288;
  float* part = (float*)(wb + o);

  auto cvt = [&](const float* s, _Float16* d, int rows, int sc, int dc, int ins) {
    int e = rows * dc;
    cvt_pad<<<(e + 255) / 256, 256, 0, stream>>>(s, d, rows, sc, dc, ins);
  };
  cvt(W[0], W0h, 256, 95, 96, -1);
  cvt(W[1], W1h, 256, 256, 256, -1);
  cvt(W[2], W2h, 256, 256, 256, -1);
  cvt(W[3], W3h, 256, 256, 256, -1);
  cvt(W[4], W4h, 256, 351, 352, 95);   // zero col inserted at skip boundary
  cvt(W[5], W5h, 256, 256, 256, -1);
  cvt(W[6], W6h, 256, 256, 256, -1);
  cvt(W[7], W7h, 256, 256, 256, -1);
  cvt(Wf, Wfh, 256, 256, 256, -1);
  cvt(Wd, Wdh, 128, 283, 288, -1);

  SP.part = part;
  mnerf_kernel<<<(N + 255) / 256, 256, 0, stream>>>(SP, N);

  BigParams BP;
  BP.x = x; BP.vid = vid; BP.code = code;
  BP.b0 = b[0]; BP.b1 = b[1]; BP.b2 = b[2]; BP.b3 = b[3];
  BP.b4 = b[4]; BP.b5 = b[5]; BP.b6 = b[6]; BP.b7 = b[7];
  BP.bf = bf; BP.bd = bd; BP.Ws = Ws; BP.bs = bs; BP.Wr = Wr; BP.br = br;
  BP.W0h = W0h; BP.W1h = W1h; BP.W2h = W2h; BP.W3h = W3h; BP.W4h = W4h;
  BP.W5h = W5h; BP.W6h = W6h; BP.W7h = W7h; BP.Wfh = Wfh; BP.Wdh = Wdh;
  BP.part_small = part;
  BP.out = (float*)d_out;

  const size_t shmem =
      (size_t)(128 * 96 + 128 * HS + 2 * WSLOT) * sizeof(_Float16);
  bignerf_kernel<<<N / 128, 256, shmem, stream>>>(BP);
}
